// ClassicalMPGNN_21930103013841
// MI455X (gfx1250) — matrix operand order repaired and checked by reading, not compiled, not run on hardware
//
#include <hip/hip_runtime.h>
#include <hip/hip_bf16.h>

// ---------------------------------------------------------------------------
// ClassicalMPGNN on MI455X (gfx1250).
// Edge MLP (128->256->256->64) as v_wmma_f32_16x16x32_f16; all three weight
// matrices DMA'd to LDS once per block via the Tensor Data Mover, reused by
// 8 waves x 5 tiles (640 edges) per block. f32 atomics for scatter/pool.
// ---------------------------------------------------------------------------

typedef _Float16 half_t;
typedef __attribute__((ext_vector_type(16))) _Float16 v16h;
typedef __attribute__((ext_vector_type(8)))  float    v8f;
typedef unsigned int u32x4 __attribute__((ext_vector_type(4)));
typedef int          i32x8 __attribute__((ext_vector_type(8)));
typedef int          i32x4 __attribute__((ext_vector_type(4)));

#define NN_NODES  50000
#define NN_EDGES  800000
#define FD        64
#define NGRAPH    500

// LDS layout (halves): W1t[256][128] | W2t[256][256] | W3t[64][256] | h[8][16][STR]
#define W1_OFF_H   0
#define W2_OFF_H   32768
#define W3_OFF_H   98304
#define WTOT_H     114688          // 229376 bytes, contiguous in workspace too
#define STR        264             // padded halves per intermediate row
#define H_OFF_H    WTOT_H
#define SMEM_HALVES (WTOT_H + 8 * 16 * STR)
#define SMEM_BYTES  (SMEM_HALVES * 2)   // 296960 B  (<= 320KB WGP LDS)

#define WAVES_PER_BLOCK 8
#define TILES_PER_WAVE  5
#define TILES_PER_BLOCK (WAVES_PER_BLOCK * TILES_PER_WAVE)   // 40
#define EDGE_BLOCKS     (NN_EDGES / (16 * TILES_PER_BLOCK))  // 1250

#if defined(__has_builtin)
#if __has_builtin(__builtin_amdgcn_tensor_load_to_lds) && \
    __has_builtin(__builtin_amdgcn_s_wait_tensorcnt)
#define USE_TDM 1
#endif
#endif

union F16x16 { v16h v; half_t h[16]; float4 q[2]; };
union F32x8  { v8f  v; float  f[8]; };

__device__ __forceinline__ v16h load_frag16(const half_t* p0, const half_t* p1) {
  F16x16 u;
  u.q[0] = *reinterpret_cast<const float4*>(p0);
  u.q[1] = *reinterpret_cast<const float4*>(p1);
  return u.v;
}

// ------------------------- small utility kernels ---------------------------

__global__ void zero_f32_kernel(float* __restrict__ p, long n) {
  long i = (long)blockIdx.x * blockDim.x + threadIdx.x;
  if (i < n) p[i] = 0.0f;
}

__global__ void cvt_f32_to_f16_kernel(const float* __restrict__ s,
                                      half_t* __restrict__ d, long n) {
  long i = (long)blockIdx.x * blockDim.x + threadIdx.x;
  if (i < n) d[i] = (half_t)s[i];
}

// Wt[n*K + k] = (f16) W[k*N + n]   (row-major [K,N] -> column-major [N,K])
__global__ void cvt_transpose_kernel(const float* __restrict__ W,
                                     half_t* __restrict__ Wt, int K, int N) {
  int i = blockIdx.x * blockDim.x + threadIdx.x;
  if (i < K * N) {
    int n = i / K, k = i % K;
    Wt[i] = (half_t)W[(long)k * N + n];
  }
}

// ------------------------------ edge MLP -----------------------------------

__global__ __launch_bounds__(256) void edge_mlp_kernel(
    const half_t* __restrict__ xh,    // [NN_NODES][64] f16
    const int*    __restrict__ ei,    // [2][NN_EDGES]
    const half_t* __restrict__ wsrc,  // contiguous W1t|W2t|W3t (229376 B)
    const float*  __restrict__ b1,    // [256]
    const float*  __restrict__ b2,    // [256]
    const float*  __restrict__ b3,    // [64]
    float*        __restrict__ node)  // [NN_NODES][64] f32 accumulator
{
  extern __shared__ half_t smem[];

  const int wave  = threadIdx.x >> 5;
  const int lane  = threadIdx.x & 31;
  const int lrow  = lane & 15;                   // A row / B column / D column
  const int lhalf = lane >> 4;                   // 0 or 1

  // ---- stage all weights into LDS once per block (TDM DMA if available) ----
#if USE_TDM
  if (threadIdx.x == 0) {
    unsigned long long ga = (unsigned long long)(uintptr_t)wsrc;
    unsigned int ldsa = (unsigned int)(uintptr_t)(&smem[W1_OFF_H]);
    u32x4 g0;
    g0[0] = 1u;                                        // count=1, user mode
    g0[1] = ldsa;                                      // lds_addr
    g0[2] = (unsigned int)ga;                          // global_addr[31:0]
    g0[3] = ((unsigned int)(ga >> 32) & 0x01FFFFFFu)   // global_addr[56:32]
            | (2u << 30);                              // type = 2 ("image")
    i32x8 g1;
    g1[0] = (int)(2u << 16);        // wg_mask=0, data_size=4B, no pad/iter
    g1[1] = (int)(4096u << 16);     // tensor_dim0[15:0]=4096 at bits 63:48
    g1[2] = (int)(14u << 16);       // tensor_dim1[15:0]=14   at bits 95:80
    g1[3] = (int)(4096u << 16);     // tile_dim0=4096         at bits 127:112
    g1[4] = 14;                     // tile_dim1=14 (tile_dim2=0)
    g1[5] = 4096;                   // tensor_dim0_stride = 4096 dwords
    g1[6] = 0;
    g1[7] = 0;
    i32x4 gz4; gz4[0] = 0; gz4[1] = 0; gz4[2] = 0; gz4[3] = 0;
    i32x8 gz8; gz8[0] = 0; gz8[1] = 0; gz8[2] = 0; gz8[3] = 0;
    gz8[4] = 0; gz8[5] = 0; gz8[6] = 0; gz8[7] = 0;
    // 6-arg form (this toolchain's clang-23 signature):
    //   (group0, group1, group2, group3, extra group, cpol)
    __builtin_amdgcn_tensor_load_to_lds(g0, g1, gz4, gz4, gz8, 0);
    __builtin_amdgcn_s_wait_tensorcnt(0);
  }
#else
  {
    float4* dst4 = (float4*)&smem[W1_OFF_H];
    const float4* src4 = (const float4*)wsrc;
    for (int i = threadIdx.x; i < (WTOT_H * 2) / 16; i += 256) dst4[i] = src4[i];
  }
#endif
  __syncthreads();

  const half_t* W1l = &smem[W1_OFF_H];
  const half_t* W2l = &smem[W2_OFF_H];
  const half_t* W3l = &smem[W3_OFF_H];
  half_t* h = &smem[H_OFF_H + wave * 16 * STR];

  v16h a[8];

  #pragma unroll 1
  for (int t = 0; t < TILES_PER_WAVE; ++t) {
    const long tile = (long)blockIdx.x * TILES_PER_BLOCK + wave * TILES_PER_WAVE + t;
    const long eb   = tile * 16;

    const int eR = ei[eb + lrow];               // source node for A row lrow
    const int eC = ei[NN_EDGES + eb + lrow];
    int rD[8];                                  // scatter targets (D rows)
    #pragma unroll
    for (int r = 0; r < 8; ++r) rD[r] = ei[eb + lhalf * 8 + r];

    // ---------------- layer 1: [16x128] @ [128x256] ----------------
    #pragma unroll
    for (int ks = 0; ks < 4; ++ks) {
      const half_t* src = (ks < 2) ? (xh + (long)eR * FD) : (xh + (long)eC * FD);
      const int o = (ks & 1) * 32 + lhalf * 8;
      a[ks] = load_frag16(src + o, src + o + 16);
    }
    #pragma unroll 1
    for (int nt = 0; nt < 16; ++nt) {
      const int n = nt * 16 + lrow;
      v8f c = {};
      #pragma unroll
      for (int ks = 0; ks < 4; ++ks) {
        const half_t* wp = W1l + (long)n * 128 + ks * 32 + lhalf * 8;
        v16h bf = load_frag16(wp, wp + 16);
        c = __builtin_amdgcn_wmma_f32_16x16x32_f16(false, a[ks], false, bf,
                                                   (short)0, c, false, false);
      }
      F32x8 cu; cu.v = c;
      const float bias = b1[n];
      #pragma unroll
      for (int r = 0; r < 8; ++r) {
        float v = cu.f[r] + bias;
        v = v > 0.0f ? v : 0.0f;
        h[(lhalf * 8 + r) * STR + n] = (half_t)v;
      }
    }

    // ---------------- layer 2: [16x256] @ [256x256] ----------------
    #pragma unroll
    for (int ks = 0; ks < 8; ++ks) {
      const half_t* p = &h[lrow * STR + ks * 32 + lhalf * 8];
      a[ks] = load_frag16(p, p + 16);           // reads precede overwrites
    }
    #pragma unroll 1
    for (int nt = 0; nt < 16; ++nt) {
      const int n = nt * 16 + lrow;
      v8f c = {};
      #pragma unroll
      for (int ks = 0; ks < 8; ++ks) {
        const half_t* wp = W2l + (long)n * 256 + ks * 32 + lhalf * 8;
        v16h bf = load_frag16(wp, wp + 16);
        c = __builtin_amdgcn_wmma_f32_16x16x32_f16(false, a[ks], false, bf,
                                                   (short)0, c, false, false);
      }
      F32x8 cu; cu.v = c;
      const float bias = b2[n];
      #pragma unroll
      for (int r = 0; r < 8; ++r) {
        float v = cu.f[r] + bias;
        v = v > 0.0f ? v : 0.0f;
        h[(lhalf * 8 + r) * STR + n] = (half_t)v;
      }
    }

    // ------------- layer 3: [16x256] @ [256x64] + scatter-add -------------
    #pragma unroll
    for (int ks = 0; ks < 8; ++ks) {
      const half_t* p = &h[lrow * STR + ks * 32 + lhalf * 8];
      a[ks] = load_frag16(p, p + 16);
    }
    #pragma unroll 1
    for (int nt = 0; nt < 4; ++nt) {
      const int n = nt * 16 + lrow;
      v8f c = {};
      #pragma unroll
      for (int ks = 0; ks < 8; ++ks) {
        const half_t* wp = W3l + (long)n * 256 + ks * 32 + lhalf * 8;
        v16h bf = load_frag16(wp, wp + 16);
        c = __builtin_amdgcn_wmma_f32_16x16x32_f16(false, a[ks], false, bf,
                                                   (short)0, c, false, false);
      }
      F32x8 cu; cu.v = c;
      const float bias = b3[n];
      #pragma unroll
      for (int r = 0; r < 8; ++r) {
        atomicAdd(&node[(long)rD[r] * FD + n], cu.f[r] + bias);
      }
    }
  }
}

// --------------------------- pooling + head --------------------------------

__global__ void pool_kernel(const float* __restrict__ node,
                            const int* __restrict__ batch,
                            float* __restrict__ pooled) {
  long i = (long)blockIdx.x * blockDim.x + threadIdx.x;
  if (i < (long)NN_NODES * FD) {
    int nidx = (int)(i >> 6);
    int d    = (int)(i & 63);
    atomicAdd(&pooled[(long)batch[nidx] * FD + d], node[i]);
  }
}

__global__ __launch_bounds__(32) void final_mlp_kernel(
    const float* __restrict__ pooled, const float* __restrict__ Wm1,
    const float* __restrict__ bm1, const float* __restrict__ Wm2,
    const float* __restrict__ bm2, float* __restrict__ out) {
  __shared__ float hid[16];
  const int g = blockIdx.x, t = threadIdx.x;
  if (t < 16) {
    float s = bm1[t];
    #pragma unroll 4
    for (int k = 0; k < FD; ++k) s += pooled[(long)g * FD + k] * Wm1[k * 16 + t];
    hid[t] = s > 0.0f ? s : 0.0f;
  }
  __syncthreads();
  if (t < 2) {
    float s = bm2[t];
    #pragma unroll
    for (int j = 0; j < 16; ++j) s += hid[j] * Wm2[j * 2 + t];
    out[g * 2 + t] = s;
  }
}

// ------------------------------- launcher ----------------------------------

extern "C" void kernel_launch(void* const* d_in, const int* in_sizes, int n_in,
                              void* d_out, int out_size, void* d_ws, size_t ws_size,
                              hipStream_t stream) {
  (void)in_sizes; (void)n_in; (void)out_size; (void)ws_size;

  const float* x   = (const float*)d_in[0];
  const int*   ei  = (const int*)d_in[1];
  const int*   bat = (const int*)d_in[2];
  const float* W1  = (const float*)d_in[3];
  const float* b1  = (const float*)d_in[4];
  const float* W2  = (const float*)d_in[5];
  const float* b2  = (const float*)d_in[6];
  const float* W3  = (const float*)d_in[7];
  const float* b3  = (const float*)d_in[8];
  const float* Wm1 = (const float*)d_in[9];
  const float* bm1 = (const float*)d_in[10];
  const float* Wm2 = (const float*)d_in[11];
  const float* bm2 = (const float*)d_in[12];
  float* out = (float*)d_out;

  char*  ws  = (char*)d_ws;
  size_t off = 0;
  auto alloc = [&](size_t bytes) -> void* {
    void* p = ws + off;
    off += (bytes + 255) & ~(size_t)255;
    return p;
  };
  half_t* xh     = (half_t*)alloc((size_t)NN_NODES * FD * 2);   // 6.4 MB
  half_t* W1t    = (half_t*)alloc((size_t)256 * 128 * 2);       // 64 KB  (contig)
  half_t* W2t    = (half_t*)alloc((size_t)256 * 256 * 2);       // 128 KB (contig)
  half_t* W3t    = (half_t*)alloc((size_t)64  * 256 * 2);       // 32 KB  (contig)
  float*  node   = (float*)alloc((size_t)NN_NODES * FD * 4);    // 12.8 MB
  float*  pooled = (float*)alloc((size_t)NGRAPH  * FD * 4);

  // 1) precision conversion / weight re-layout (f32 -> f16, W -> W^T)
  {
    long n = (long)NN_NODES * FD;
    cvt_f32_to_f16_kernel<<<(int)((n + 255) / 256), 256, 0, stream>>>(x, xh, n);
  }
  cvt_transpose_kernel<<<(128 * 256) / 256, 256, 0, stream>>>(W1, W1t, 128, 256);
  cvt_transpose_kernel<<<(256 * 256) / 256, 256, 0, stream>>>(W2, W2t, 256, 256);
  cvt_transpose_kernel<<<(256 * 64)  / 256, 256, 0, stream>>>(W3, W3t, 256, 64);

  // 2) zero accumulators (every launch; harness does not re-poison)
  {
    long n = (long)NN_NODES * FD;
    zero_f32_kernel<<<(int)((n + 255) / 256), 256, 0, stream>>>(node, n);
    long m = (long)NGRAPH * FD;
    zero_f32_kernel<<<(int)((m + 255) / 256), 256, 0, stream>>>(pooled, m);
  }

  // 3) edge MLP: TDM-staged weights in LDS, 296960 B dynamic LDS per block,
  //    8 waves x 5 tiles = 640 edges/block, 1250 blocks.
  edge_mlp_kernel<<<EDGE_BLOCKS, 256, SMEM_BYTES, stream>>>(
      xh, ei, W1t, b1, b2, b3, node);

  // 4) global_add_pool over batch
  {
    long n = (long)NN_NODES * FD;
    pool_kernel<<<(int)((n + 255) / 256), 256, 0, stream>>>(node, bat, pooled);
  }

  // 5) head MLP -> [500, 2]
  final_mlp_kernel<<<NGRAPH, 32, 0, stream>>>(pooled, Wm1, bm1, Wm2, bm2, out);
}
